// OptimizedRNN2D_66571993088932
// MI455X (gfx1250) — matrix operand order repaired and checked by reading, not compiled, not run on hardware
//
#include <hip/hip_runtime.h>
#include <stdint.h>

// ---------------------------------------------------------------------------
// Fused 2D-RNN for MI455X (gfx1250, wave32, WMMA).
//   grid = B * (W/16) = 128 workgroups; block = 256 threads = 8 waves.
//   Each workgroup owns 16 image columns (fixed b, w0..w0+15) and sweeps all
//   H=256 rows sequentially; h-state (16x128 bf16) lives in LDS.
//   Wave w computes hid channels [16w,16w+16); waves 0..3 also do the 1x1 conv.
//   x row tiles (16x128 f32, contiguous 8KB) are double-buffered into LDS by
//   the Tensor Data Mover (tensor_load_to_lds), synchronized via TENSORcnt.
// ---------------------------------------------------------------------------

#define B_    8
#define H_    256
#define W_    256
#define DIN   128
#define DHID  128
#define DOUT  64
#define HW    (H_ * W_)
#define HID_OFF ((size_t)B_ * DOUT * HW)   // out tensor precedes hid tensor

typedef __attribute__((ext_vector_type(16))) __bf16 v16bf;
typedef __attribute__((ext_vector_type(8)))  __bf16 v8bf;
typedef __attribute__((ext_vector_type(8)))  float  v8f;
typedef __attribute__((ext_vector_type(4)))  float  f32x4;
typedef __attribute__((ext_vector_type(4)))  unsigned int u32x4;
typedef __attribute__((ext_vector_type(4)))  int    i32x4;
typedef __attribute__((ext_vector_type(8)))  int    i32x8;

// A-operand (16x32 bf16): lane l holds row M=l%16 and 16 contiguous K
// elements at kb (kb = 32*kk + (l<16?0:16)), matching the B fill.
static __device__ __forceinline__ v16bf load_a(const __bf16* p) {
    union { v16bf v; v8bf h[2]; } u;
    u.h[0] = *(const v8bf*)(p);
    u.h[1] = *(const v8bf*)(p + 8);
    return u.v;
}

static __device__ __forceinline__ unsigned lds_off_of(const void* p) {
    // generic (flat) pointers to LDS carry the byte offset in the low 32 bits
    return (unsigned)(uintptr_t)p;
}

// Issue one TDM copy: 2048 contiguous f32 (8KB) global -> LDS.
static __device__ __forceinline__ void tdm_issue(const void* gsrc, unsigned lds_addr) {
    unsigned long long ga = (unsigned long long)(uintptr_t)gsrc;
    u32x4 g0;
    g0[0] = 1u;                                         // count=1 (valid user D#)
    g0[1] = lds_addr;                                   // lds_addr (bytes)
    g0[2] = (unsigned)(ga & 0xffffffffull);             // global_addr[31:0]
    g0[3] = (unsigned)((ga >> 32) & 0x01ffffffull)      // global_addr[56:32]
          | (2u << 30);                                 // type=2 ("image")
    i32x8 g1;
    g1[0] = (2 << 16);            // workgroup_mask=0, data_size=2 (4-byte)
    g1[1] = (int)(2048u << 16);   // tensor_dim0 = 2048 (low 16 bits -> [31:16])
    g1[2] = (int)(1u << 16);      // tensor_dim0 hi = 0; tensor_dim1 = 1
    g1[3] = (int)(2048u << 16);   // tensor_dim1 hi = 0; tile_dim0 = 2048
    g1[4] = 1;                    // tile_dim1 = 1; tile_dim2 = 0
    g1[5] = 2048;                 // tensor_dim0_stride (low 32)
    g1[6] = 0;
    g1[7] = 0;
    i32x4 z4 = {0, 0, 0, 0};      // groups 2/3 unused (<=2D tensor)
#if defined(__clang_major__) && (__clang_major__ >= 23)
    i32x8 z8 = {0, 0, 0, 0, 0, 0, 0, 0};
    __builtin_amdgcn_tensor_load_to_lds(g0, g1, z4, z4, z8, 0);
#else
    __builtin_amdgcn_tensor_load_to_lds(g0, g1, z4, z4, 0);
#endif
}

static __device__ __forceinline__ v8f wmma_bf16(v16bf a, v16bf b, v8f c) {
    return __builtin_amdgcn_wmma_f32_16x16x32_bf16(
        /*neg_a=*/false, a, /*neg_b=*/false, b,
        /*c_mod=*/(short)0, c, /*reuse_a=*/false, /*reuse_b=*/false);
}

__global__ __launch_bounds__(256) void rnn2d_fused_kernel(
    const float* __restrict__ x,
    const float* __restrict__ Wih,  const float* __restrict__ bih,
    const float* __restrict__ Wtop, const float* __restrict__ btop,
    const float* __restrict__ bleft,
    const float* __restrict__ Wcv,  const float* __restrict__ bcv,
    float* __restrict__ out)
{
    __shared__ __align__(16) float  xf[2][16 * DIN];   // TDM double buffer (f32)
    __shared__ __align__(32) __bf16 xb[16 * DIN];      // current row, bf16
    __shared__ __align__(32) __bf16 hb[16 * DHID];     // h state [m][ch], bf16

    const int tid  = threadIdx.x;
    const int wave = tid >> 5;
    const int lane = tid & 31;
    const int n    = lane & 15;        // N index within 16x16 tile
    const int hi   = lane >> 4;        // lane half (0/1)
    const int m    = n;                // A-matrix row index for this lane
    const int b    = blockIdx.x >> 4;
    const int w0   = (blockIdx.x & 15) << 4;
    const int j    = wave * 16 + n;    // hid channel owned by this lane

    // ---- loop-invariant weights in WMMA B-layout (lane=N, contiguous K16) ----
    v16bf Bih[4], Btop[4];
    v16bf Bcv[4] = {};
#pragma unroll
    for (int kk = 0; kk < 4; ++kk) {
        const int kb = kk * 32 + hi * 16;
        const float* pih = Wih  + j * DIN  + kb;
        const float* ptp = Wtop + j * DHID + kb;
#pragma unroll
        for (int t = 0; t < 16; ++t) {
            Bih[kk][t]  = (__bf16)pih[t];
            Btop[kk][t] = (__bf16)ptp[t];
        }
    }
    if (wave < 4) {
#pragma unroll
        for (int kk = 0; kk < 4; ++kk) {
            const int kb = kk * 32 + hi * 16;
            const float* pcv = Wcv + (wave * 16 + n) * DHID + kb;
#pragma unroll
            for (int t = 0; t < 16; ++t) Bcv[kk][t] = (__bf16)pcv[t];
        }
    }
    const float bias_h = bih[j] + bleft[j] + btop[j];   // folded per-row bias
    const float bias_c = (wave < 4) ? bcv[wave * 16 + n] : 0.f;

    // ---- h0 = 0 ----
    for (int t = tid; t < 16 * DHID; t += 256) hb[t] = (__bf16)0.f;

    // ---- prologue: TDM fetch of row 0 ----
    if (wave == 0)
        tdm_issue(x + ((size_t)b * H_ * W_ + (size_t)w0) * DIN, lds_off_of(&xf[0][0]));
    __syncthreads();

    for (int i = 0; i < H_; ++i) {
        if (wave == 0) __builtin_amdgcn_s_wait_tensorcnt(0);
        __syncthreads();                                  // row i tile resident

        // convert current x tile f32 -> bf16 once (shared by all 8 waves)
        {
            const float* src = xf[i & 1];
#pragma unroll
            for (int t = 0; t < 8; ++t)
                xb[tid * 8 + t] = (__bf16)src[tid * 8 + t];
        }
        // prefetch next row via TDM into the other buffer (overlaps compute)
        if (wave == 0 && (i + 1) < H_)
            tdm_issue(x + (((size_t)b * H_ + (i + 1)) * W_ + w0) * DIN,
                      lds_off_of(&xf[(i + 1) & 1][0]));
        __syncthreads();                                  // xb ready

        // h_new = relu( x@Wih^T + h_prev@Wtop^T + bias )
        v8f acc = {bias_h, bias_h, bias_h, bias_h, bias_h, bias_h, bias_h, bias_h};
#pragma unroll
        for (int kk = 0; kk < 4; ++kk) {
            const int kb = kk * 32 + hi * 16;
            acc = wmma_bf16(load_a(&xb[m * DIN  + kb]), Bih[kk],  acc);
            acc = wmma_bf16(load_a(&hb[m * DHID + kb]), Btop[kk], acc);
        }
#pragma unroll
        for (int t = 0; t < 8; ++t) acc[t] = fmaxf(acc[t], 0.f);

        __syncthreads();                                  // all done reading old hb

        // publish new h to LDS (bf16) and to global (f32, NCHW hid output)
#pragma unroll
        for (int t = 0; t < 8; ++t)
            hb[(t + hi * 8) * DHID + j] = (__bf16)acc[t];
        {
            float* p = out + HID_OFF + ((size_t)(b * DHID + j)) * HW
                     + (size_t)i * W_ + (w0 + hi * 8);
            f32x4 s0 = {acc[0], acc[1], acc[2], acc[3]};
            f32x4 s1 = {acc[4], acc[5], acc[6], acc[7]};
            *(f32x4*)(p)     = s0;
            *(f32x4*)(p + 4) = s1;
        }
        __syncthreads();                                  // new hb visible

        // 1x1 conv: out = relu( h@Wcv^T + b_conv ), waves 0..3 (64 channels)
        if (wave < 4) {
            v8f co = {bias_c, bias_c, bias_c, bias_c, bias_c, bias_c, bias_c, bias_c};
#pragma unroll
            for (int kk = 0; kk < 4; ++kk) {
                const int kb = kk * 32 + hi * 16;
                co = wmma_bf16(load_a(&hb[m * DHID + kb]), Bcv[kk], co);
            }
#pragma unroll
            for (int t = 0; t < 8; ++t) co[t] = fmaxf(co[t], 0.f);
            float* p = out + ((size_t)(b * DOUT + wave * 16 + n)) * HW
                     + (size_t)i * W_ + (w0 + hi * 8);
            f32x4 s0 = {co[0], co[1], co[2], co[3]};
            f32x4 s1 = {co[4], co[5], co[6], co[7]};
            *(f32x4*)(p)     = s0;
            *(f32x4*)(p + 4) = s1;
        }
    }
}

extern "C" void kernel_launch(void* const* d_in, const int* in_sizes, int n_in,
                              void* d_out, int out_size, void* d_ws, size_t ws_size,
                              hipStream_t stream) {
    (void)in_sizes; (void)n_in; (void)out_size; (void)d_ws; (void)ws_size;
    const float* x     = (const float*)d_in[0];
    const float* W_ih  = (const float*)d_in[1];
    const float* b_ih  = (const float*)d_in[2];
    const float* W_top = (const float*)d_in[3];
    const float* b_top = (const float*)d_in[4];
    // d_in[5] = W_left: unused — reference reads a never-written row, so only
    // b_left contributes.
    const float* b_left = (const float*)d_in[6];
    const float* W_conv = (const float*)d_in[7];
    const float* b_conv = (const float*)d_in[8];

    dim3 grid(B_ * (W_ / 16));   // 128 workgroups
    dim3 block(256);             // 8 wave32s
    rnn2d_fused_kernel<<<grid, block, 0, stream>>>(
        x, W_ih, b_ih, W_top, b_top, b_left, W_conv, b_conv, (float*)d_out);
}
